// AnomalyAwareMemory_13924283974195
// MI455X (gfx1250) — matrix operand order repaired and checked by reading, not compiled, not run on hardware
//
#include <hip/hip_runtime.h>
#include <hip/hip_bf16.h>

// ---------------- problem constants ----------------
#define DN   16384   // N embeddings
#define DK   4096    // K memory slots
#define DD   128     // D features
#define DNC  16384   // num classes
#define MOM  0.01f
#define QSCALE 0.88388347648f   // 1/(sqrt(128)*0.1)
#define NCH  (DK / 32)          // 32-key chunks in attention

// ---------------- vector types / helpers ----------------
typedef __attribute__((ext_vector_type(8)))  float        v8f;
typedef __attribute__((ext_vector_type(16))) __bf16       v16bf;
typedef __attribute__((ext_vector_type(4)))  unsigned int v4u;

union Frag { v16bf v; v4u q[2]; };

__device__ __forceinline__ unsigned short f2bf(float f) {
  unsigned u = __float_as_uint(f);
  u += 0x7fffu + ((u >> 16) & 1u);          // round-to-nearest-even
  return (unsigned short)(u >> 16);
}

// 16B async copy global -> LDS, tracked by ASYNCcnt (CDNA5 GLOBAL_LOAD_ASYNC_TO_LDS_B128)
__device__ __forceinline__ void async_copy16(unsigned lds_off, const void* gptr) {
  asm volatile("global_load_async_to_lds_b128 %0, %1, off"
               :: "v"(lds_off), "v"((unsigned long long)(size_t)gptr)
               : "memory");
}

// ---------------- workspace layout (bytes) ----------------
constexpr size_t OFF_MEAN  = 0;          // 128 f32
constexpr size_t OFF_RM    = 512;        // 128 f32
constexpr size_t OFF_A     = 1024;       // 128x128 f32  (matrix to invert)
constexpr size_t OFF_INV   = 66560;      // 128x128 f32
constexpr size_t OFF_SCAL  = 132096;     // scalars: [0]=minbits [1]=maxbits [2]=unique
constexpr size_t OFF_CNT   = 132352;     // 16384 i32
constexpr size_t OFF_MAHAL = 197888;     // 16384 f32
constexpr size_t OFF_KEY   = 263424;     // 16384 f32 (importance, sorted desc)
constexpr size_t OFF_IDX   = 328960;     // 16384 i32 (argsort indices)
constexpr size_t OFF_MEM   = 394496;     // 4096x128 f32 (updated memory)
constexpr size_t OFF_QB    = 2491648;    // 16384x128 bf16
constexpr size_t OFF_KB    = 6685952;    // 4096x128 bf16
constexpr size_t OFF_VT    = 7734528;    // 128x4096 bf16 (transposed V, incl bias)

// ================= init: counts + scalars =================
__global__ void k_init(int* counts, unsigned int* scal) {
  int i = blockIdx.x * blockDim.x + threadIdx.x;
  if (i < DNC) counts[i] = 0;
  if (i == 0) { scal[0] = 0x7f800000u; scal[1] = 0u; scal[2] = 0u; }
}

// ================= column mean + running mean =================
__global__ void k_mean(const float* __restrict__ emb, const float* __restrict__ rmean,
                       float* __restrict__ mean, float* __restrict__ rm) {
  __shared__ float red[256];
  const int d = blockIdx.x, tid = threadIdx.x;
  float acc = 0.f;
  for (int n = tid; n < DN; n += 256) acc += emb[(size_t)n * DD + d];
  red[tid] = acc; __syncthreads();
  for (int s = 128; s > 0; s >>= 1) { if (tid < s) red[tid] += red[tid + s]; __syncthreads(); }
  if (tid == 0) {
    float m = red[0] / (float)DN;
    mean[d] = m;
    rm[d] = (1.f - MOM) * rmean[d] + MOM * m;
  }
}

// ================= covariance -> A = rc + 1e-6 I ; INV = I =================
__global__ void k_cov(const float* __restrict__ emb, const float* __restrict__ mean,
                      const float* __restrict__ rcov, float* __restrict__ A,
                      float* __restrict__ INV) {
  __shared__ float Ei[128][16];
  __shared__ float Ej[128][16];
  const int tid = threadIdx.x;
  const int bi = blockIdx.x >> 3, bj = blockIdx.x & 7;
  const int ti = tid >> 4, tj = tid & 15;
  float acc = 0.f;
  for (int n0 = 0; n0 < DN; n0 += 128) {
    #pragma unroll
    for (int t = 0; t < 8; ++t) {
      int idx = tid + t * 256;
      int r = idx >> 4, c = idx & 15;
      Ei[r][c] = emb[(size_t)(n0 + r) * DD + bi * 16 + c] - mean[bi * 16 + c];
      Ej[r][c] = emb[(size_t)(n0 + r) * DD + bj * 16 + c] - mean[bj * 16 + c];
    }
    __syncthreads();
    #pragma unroll 8
    for (int r = 0; r < 128; ++r) acc += Ei[r][ti] * Ej[r][tj];
    __syncthreads();
  }
  const int i = bi * 16 + ti, j = bj * 16 + tj;
  const size_t ij = (size_t)i * DD + j;
  float cov = acc / (float)(DN - 1);
  A[ij] = (1.f - MOM) * rcov[ij] + MOM * cov + ((i == j) ? 1e-6f : 0.f);
  INV[ij] = (i == j) ? 1.f : 0.f;
}

// ================= Gauss-Jordan inverse (SPD, no pivoting) =================
__global__ void k_inverse(float* __restrict__ A, float* __restrict__ INV) {
  __shared__ float fcol[128];
  const int tid = threadIdx.x;                 // 256 threads
  for (int p = 0; p < 128; ++p) {
    if (tid < 128) fcol[tid] = A[tid * 128 + p];
    __syncthreads();
    float pinv = 1.0f / fcol[p];
    if (tid < 128) A[p * 128 + tid] *= pinv;
    else           INV[p * 128 + (tid - 128)] *= pinv;
    __syncthreads();
    for (int idx = tid; idx < 128 * 256; idx += 256) {
      int r = idx >> 8, c = idx & 255;
      if (r != p) {
        float f = fcol[r];
        if (c < 128) A[r * 128 + c]           -= f * A[p * 128 + c];
        else         INV[r * 128 + (c - 128)] -= f * INV[p * 128 + (c - 128)];
      }
    }
    __syncthreads();
  }
}

// ================= Mahalanobis distance + global min/max =================
__global__ void k_mahal(const float* __restrict__ emb, const float* __restrict__ rm,
                        const float* __restrict__ INV, float* __restrict__ mahal,
                        unsigned int* __restrict__ scal) {
  __shared__ float c[128];
  __shared__ float red[128];
  const int n = blockIdx.x, tid = threadIdx.x;
  c[tid] = emb[(size_t)n * DD + tid] - rm[tid];
  __syncthreads();
  float acc = 0.f;
  #pragma unroll 8
  for (int j = 0; j < 128; ++j) acc += c[j] * INV[j * 128 + tid];
  red[tid] = acc * c[tid]; __syncthreads();
  for (int s = 64; s > 0; s >>= 1) { if (tid < s) red[tid] += red[tid + s]; __syncthreads(); }
  if (tid == 0) {
    float m = sqrtf(red[0] + 1e-8f);
    mahal[n] = m;
    unsigned b = __float_as_uint(m);           // m >= 0 -> uint order == float order
    atomicMin(&scal[0], b);
    atomicMax(&scal[1], b);
  }
}

// ================= label bincount / unique count =================
__global__ void k_count(const int* __restrict__ labels, int* __restrict__ counts) {
  int n = blockIdx.x * blockDim.x + threadIdx.x;
  if (n < DN) atomicAdd(&counts[labels[n]], 1);
}
__global__ void k_unique(const int* __restrict__ counts, unsigned int* __restrict__ scal) {
  int i = blockIdx.x * blockDim.x + threadIdx.x;
  if (i < DNC && counts[i] > 0) atomicAdd(&scal[2], 1u);
}

// ================= importance + sort keys =================
__global__ void k_importance(const float* __restrict__ mahal, const unsigned int* __restrict__ scal,
                             float* __restrict__ key, int* __restrict__ idx) {
  int n = blockIdx.x * blockDim.x + threadIdx.x;
  if (n >= DN) return;
  float mn = __uint_as_float(scal[0]);
  float mx = __uint_as_float(scal[1]);
  float ur = (float)scal[2] / (float)DN;
  float an = (mahal[n] - mn) / (mx - mn + 1e-8f);
  key[n] = (1.f + an) * ur;
  idx[n] = n;
}

// ================= copy memory -> workspace =================
__global__ void k_copymem(const float* __restrict__ src, float* __restrict__ dst) {
  int i = blockIdx.x * blockDim.x + threadIdx.x;
  if (i < DK * DD) dst[i] = src[i];
}

// ================= single-block bitonic sort, descending =================
__global__ void k_sort(float* __restrict__ key, int* __restrict__ idx) {
  const int tid = threadIdx.x;                 // 1024 threads
  for (unsigned k = 2; k <= DN; k <<= 1) {
    for (unsigned j = k >> 1; j > 0; j >>= 1) {
      for (unsigned i = tid; i < DN; i += 1024) {
        unsigned ixj = i ^ j;
        if (ixj > i) {
          bool up = ((i & k) == 0);            // descending overall
          float a = key[i], b = key[ixj];
          if ((a < b) == up) {
            key[i] = b; key[ixj] = a;
            int t = idx[i]; idx[i] = idx[ixj]; idx[ixj] = t;
          }
        }
      }
      __syncthreads();
    }
  }
}

// ================= sequential replacement with LDS tournament tree =================
__global__ void k_scan(const float* __restrict__ emb, const float* __restrict__ mw,
                       const float* __restrict__ key, const int* __restrict__ idx,
                       float* __restrict__ mem) {
  __shared__ float          tval[8192];
  __shared__ unsigned short tnid[8192];
  __shared__ float sImp;
  __shared__ int   sSrc, sMi, sGo;
  const int tid = threadIdx.x;                 // 256 threads
  for (int i = tid; i < DK; i += 256) { tval[4096 + i] = mw[i]; tnid[4096 + i] = (unsigned short)i; }
  __syncthreads();
  for (int level = 2048; level >= 1; level >>= 1) {
    for (int i = tid; i < level; i += 256) {
      int n = level + i;
      float lv = tval[2 * n], rv = tval[2 * n + 1];
      bool tl = lv <= rv;                      // tie -> lower index (argmin semantics)
      tval[n] = tl ? lv : rv;
      tnid[n] = tl ? tnid[2 * n] : tnid[2 * n + 1];
    }
    __syncthreads();
  }
  for (int i = 0; i < DN; ++i) {
    if (tid == 0) {
      float imp = key[i];
      sGo = (imp > tval[1]) ? 1 : 0;           // min non-decreasing, imp non-increasing
      sImp = imp; sSrc = idx[i]; sMi = (int)tnid[1];
    }
    __syncthreads();
    if (!sGo) break;                           // uniform: all remaining steps are no-ops
    int mi = sMi, src = sSrc;
    if (tid < 128) mem[(size_t)mi * DD + tid] = emb[(size_t)src * DD + tid];
    if (tid == 0) {
      tval[4096 + mi] = sImp;
      int node = (4096 + mi) >> 1;
      while (node >= 1) {
        int l = node * 2, r = l + 1;
        float lv = tval[l], rv = tval[r];
        bool tl = lv <= rv;
        tval[node] = tl ? lv : rv;
        tnid[node] = tl ? tnid[l] : tnid[r];
        node >>= 1;
      }
    }
    __syncthreads();
  }
}

// ================= projections -> bf16 =================
__global__ void k_projQ(const float* __restrict__ emb, const float* __restrict__ Wq,
                        const float* __restrict__ bq, unsigned short* __restrict__ Qb) {
  __shared__ float c[128];
  const int n = blockIdx.x, tid = threadIdx.x;
  c[tid] = emb[(size_t)n * DD + tid]; __syncthreads();
  float acc = bq[tid];
  #pragma unroll 8
  for (int j = 0; j < 128; ++j) acc += c[j] * Wq[tid * 128 + j];
  Qb[(size_t)n * DD + tid] = f2bf(acc * QSCALE);
}

__global__ void k_projKV(const float* __restrict__ mem,
                         const float* __restrict__ Wk, const float* __restrict__ bk,
                         const float* __restrict__ Wv, const float* __restrict__ bv,
                         unsigned short* __restrict__ Kb, unsigned short* __restrict__ Vt) {
  __shared__ float c[128];
  const int k = blockIdx.x, tid = threadIdx.x;
  c[tid] = mem[(size_t)k * DD + tid]; __syncthreads();
  float ak = bk[tid], av = bv[tid];
  #pragma unroll 8
  for (int j = 0; j < 128; ++j) { float cj = c[j]; ak += cj * Wk[tid * 128 + j]; av += cj * Wv[tid * 128 + j]; }
  Kb[(size_t)k * DD + tid] = f2bf(ak);
  Vt[(size_t)tid * DK + k] = f2bf(av);         // transposed: feature-major for PV B-frags
}

// ================= fused flash attention, bf16 WMMA + async LDS staging =================
// block = 128 threads = 4 waves; block stages each 32-key K/V chunk ONCE into LDS
// (double-buffered via GLOBAL_LOAD_ASYNC_TO_LDS_B128 / ASYNCcnt); each wave owns 16 query rows.
__global__ __launch_bounds__(128) void k_attn(const float* __restrict__ emb,
                                              const unsigned int* __restrict__ Qb,
                                              const unsigned int* __restrict__ Kb,
                                              const unsigned int* __restrict__ Vt,
                                              float* __restrict__ out) {
  __shared__ __align__(16) unsigned short Ksh[2][32 * 128];   // 2 x 8KB  key rows x features
  __shared__ __align__(16) unsigned short Vsh[2][128 * 32];   // 2 x 8KB  feature rows x keys
  __shared__ __align__(16) unsigned short Psh[4][16 * 32];    // per-wave P relayout
  const int tid = threadIdx.x;
  const int lane = tid & 31, wv = tid >> 5;
  const int rA = lane & 15, kh = lane >> 4;
  const int qbase = blockIdx.x * 64 + wv * 16;

  const v8f vzero = {0.f, 0.f, 0.f, 0.f, 0.f, 0.f, 0.f, 0.f};

  // cooperative async stage of one 32-key chunk (8 x 16B per thread)
  auto stage = [&](int j0, int buf) {
    const char* gK = (const char*)Kb + (size_t)j0 * 256;      // 32 rows x 256B, linear
    unsigned lK = (unsigned)(size_t)&Ksh[buf][0];
    #pragma unroll
    for (int i = 0; i < 4; ++i) {
      int u = tid + i * 128;                                   // 0..511
      async_copy16(lK + u * 16, gK + (size_t)u * 16);
    }
    const char* gV = (const char*)Vt;
    unsigned lV = (unsigned)(size_t)&Vsh[buf][0];
    #pragma unroll
    for (int i = 0; i < 4; ++i) {
      int u = tid + i * 128;
      int f = u >> 2, q = u & 3;                               // 64B per feature row
      async_copy16(lV + u * 16, gV + ((size_t)f * DK + j0) * 2 + q * 16);
    }
  };

  // ---- preload Q fragments (16x128 bf16 -> 4 A-frags), register-resident ----
  Frag qf[4];
  const v4u* qrow = (const v4u*)(Qb + (size_t)(qbase + rA) * 64);
  #pragma unroll
  for (int s = 0; s < 4; ++s) { qf[s].q[0] = qrow[s * 4 + kh * 2]; qf[s].q[1] = qrow[s * 4 + kh * 2 + 1]; }

  v8f O[8];
  float rmax[8], rsum[8];
  #pragma unroll
  for (int c = 0; c < 8; ++c) O[c] = vzero;
  #pragma unroll
  for (int g = 0; g < 8; ++g) { rmax[g] = -3.0e38f; rsum[g] = 0.f; }

  stage(0, 0);                                                 // prologue
  for (int j = 0; j < NCH; ++j) {
    const int buf = j & 1;
    if (j + 1 < NCH) {
      stage((j + 1) * 32, buf ^ 1);                            // overlap next copy with compute
      asm volatile("s_wait_asynccnt 0x8" ::: "memory");        // current chunk's 8 done
    } else {
      asm volatile("s_wait_asynccnt 0x0" ::: "memory");
    }
    __syncthreads();                                           // staged data visible to all waves

    // ---- all 8 K B-frags from LDS first, then back-to-back WMMAs (2 indep chains) ----
    const v4u* kp = (const v4u*)&Ksh[buf][0];
    Frag kf[8];
    #pragma unroll
    for (int g = 0; g < 2; ++g)
      #pragma unroll
      for (int s = 0; s < 4; ++s) {
        int row = g * 16 + rA;
        kf[g * 4 + s].q[0] = kp[row * 16 + s * 4 + kh * 2];
        kf[g * 4 + s].q[1] = kp[row * 16 + s * 4 + kh * 2 + 1];
      }
    v8f S0 = vzero, S1 = vzero;
    #pragma unroll
    for (int s = 0; s < 4; ++s) {
      S0 = __builtin_amdgcn_wmma_f32_16x16x32_bf16(false, qf[s].v, false, kf[s].v,     (short)0, S0, false, false);
      S1 = __builtin_amdgcn_wmma_f32_16x16x32_bf16(false, qf[s].v, false, kf[4 + s].v, (short)0, S1, false, false);
    }

    // ---- V B-frags issued early so ds latency hides under softmax VALU ----
    const v4u* vp = (const v4u*)&Vsh[buf][0];
    Frag vf[8];
    #pragma unroll
    for (int c = 0; c < 8; ++c) {
      int f = c * 16 + rA;
      vf[c].q[0] = vp[f * 4 + kh * 2];
      vf[c].q[1] = vp[f * 4 + kh * 2 + 1];
    }

    // ---- online softmax: row m = g + 8*kh lives in element g, lane-local stats ----
    #pragma unroll
    for (int g = 0; g < 8; ++g) {
      float s0 = S0[g], s1 = S1[g];
      float mx = fmaxf(s0, s1);
      #pragma unroll
      for (int off = 8; off; off >>= 1) mx = fmaxf(mx, __shfl_xor(mx, off, 32));
      float nm = fmaxf(rmax[g], mx);
      float corr = __expf(rmax[g] - nm);
      float p0 = __expf(s0 - nm), p1 = __expf(s1 - nm);
      float ss = p0 + p1;
      #pragma unroll
      for (int off = 8; off; off >>= 1) ss += __shfl_xor(ss, off, 32);
      rsum[g] = rsum[g] * corr + ss;
      rmax[g] = nm;
      #pragma unroll
      for (int c = 0; c < 8; ++c) O[c][g] *= corr;
      int m = g + 8 * kh;
      Psh[wv][m * 32 + rA]      = f2bf(p0);
      Psh[wv][m * 32 + 16 + rA] = f2bf(p1);
    }
    __syncthreads();
    // ---- relayout P (C-layout -> A-layout) via LDS, then O += P @ V ----
    Frag pf;
    const v4u* pp = (const v4u*)&Psh[wv][0];
    pf.q[0] = pp[rA * 4 + kh * 2];
    pf.q[1] = pp[rA * 4 + kh * 2 + 1];
    #pragma unroll
    for (int c = 0; c < 8; ++c)
      O[c] = __builtin_amdgcn_wmma_f32_16x16x32_bf16(false, pf.v, false, vf[c].v, (short)0, O[c], false, false);
  }
  // ---- out = emb + 0.5 * O / rowsum ----
  #pragma unroll
  for (int g = 0; g < 8; ++g) {
    int q = qbase + g + 8 * kh;
    float inv = 1.0f / rsum[g];
    #pragma unroll
    for (int c = 0; c < 8; ++c) {
      int d = c * 16 + rA;
      size_t o = (size_t)q * DD + d;
      out[o] = emb[o] + 0.5f * O[c][g] * inv;
    }
  }
}

// ================= launcher =================
extern "C" void kernel_launch(void* const* d_in, const int* in_sizes, int n_in,
                              void* d_out, int out_size, void* d_ws, size_t ws_size,
                              hipStream_t stream) {
  (void)in_sizes; (void)n_in; (void)out_size; (void)ws_size;
  const float* emb     = (const float*)d_in[0];
  const int*   labels  = (const int*)  d_in[1];
  const float* memory  = (const float*)d_in[2];
  const float* mweight = (const float*)d_in[3];
  // d_in[4] memory_labels: does not affect the output
  const float* rmean   = (const float*)d_in[5];
  const float* rcov    = (const float*)d_in[6];
  const float* Wq = (const float*)d_in[7];  const float* bq = (const float*)d_in[8];
  const float* Wk = (const float*)d_in[9];  const float* bk = (const float*)d_in[10];
  const float* Wv = (const float*)d_in[11]; const float* bv = (const float*)d_in[12];
  float* out = (float*)d_out;
  char*  ws  = (char*)d_ws;

  float*          mean  = (float*)(ws + OFF_MEAN);
  float*          rm    = (float*)(ws + OFF_RM);
  float*          A     = (float*)(ws + OFF_A);
  float*          INV   = (float*)(ws + OFF_INV);
  unsigned int*   scal  = (unsigned int*)(ws + OFF_SCAL);
  int*            cnts  = (int*)(ws + OFF_CNT);
  float*          mahal = (float*)(ws + OFF_MAHAL);
  float*          key   = (float*)(ws + OFF_KEY);
  int*            idx   = (int*)(ws + OFF_IDX);
  float*          mem   = (float*)(ws + OFF_MEM);
  unsigned short* Qb    = (unsigned short*)(ws + OFF_QB);
  unsigned short* Kb    = (unsigned short*)(ws + OFF_KB);
  unsigned short* Vt    = (unsigned short*)(ws + OFF_VT);

  k_init      <<<dim3(DNC / 256), dim3(256), 0, stream>>>(cnts, scal);
  k_mean      <<<dim3(DD),        dim3(256), 0, stream>>>(emb, rmean, mean, rm);
  k_cov       <<<dim3(64),        dim3(256), 0, stream>>>(emb, mean, rcov, A, INV);
  k_inverse   <<<dim3(1),         dim3(256), 0, stream>>>(A, INV);
  k_mahal     <<<dim3(DN),        dim3(128), 0, stream>>>(emb, rm, INV, mahal, scal);
  k_count     <<<dim3(DN / 256),  dim3(256), 0, stream>>>(labels, cnts);
  k_unique    <<<dim3(DNC / 256), dim3(256), 0, stream>>>(cnts, scal);
  k_importance<<<dim3(DN / 256),  dim3(256), 0, stream>>>(mahal, scal, key, idx);
  k_copymem   <<<dim3(DK * DD / 256), dim3(256), 0, stream>>>(memory, mem);
  k_sort      <<<dim3(1),         dim3(1024), 0, stream>>>(key, idx);
  k_scan      <<<dim3(1),         dim3(256), 0, stream>>>(emb, mweight, key, idx, mem);
  k_projQ     <<<dim3(DN),        dim3(128), 0, stream>>>(emb, Wq, bq, Qb);
  k_projKV    <<<dim3(DK),        dim3(128), 0, stream>>>(mem, Wk, bk, Wv, bv, Kb, Vt);
  k_attn      <<<dim3(DN / 64),   dim3(128), 0, stream>>>(emb, (const unsigned int*)Qb,
                                                          (const unsigned int*)Kb,
                                                          (const unsigned int*)Vt, out);
}
